// NonLocal2d_68513318306032
// MI455X (gfx1250) — matrix operand order repaired and checked by reading, not compiled, not run on hardware
//
#include <hip/hip_runtime.h>
#include <math.h>

// Problem constants (from reference): B=8, C=512, P=256, H=W=64 -> N=4096
#define B_ 8
#define C_ 512
#define P_ 256
#define N_ 4096
#define SCALE_ 0.0625f   // 1/sqrt(P)=1/16

typedef __attribute__((ext_vector_type(16))) __bf16 v16bf;
typedef __attribute__((ext_vector_type(8)))  float  v8f;

struct Frag32 { uint4 lo, hi; };   // 32-byte container -> v16bf
struct Pack16 { __bf16 v[8]; };    // 16-byte container -> uint4

// Load a v16bf fragment from two 16B-aligned chunks (elements 0-7 at p0, 8-15 at p1)
static __device__ __forceinline__ v16bf load_frag2(const __bf16* p0, const __bf16* p1) {
    Frag32 f;
    f.lo = *reinterpret_cast<const uint4*>(p0);
    f.hi = *reinterpret_cast<const uint4*>(p1);
    return __builtin_bit_cast(v16bf, f);
}
// Load a v16bf fragment from one contiguous 32B run
static __device__ __forceinline__ v16bf load_frag_g(const __bf16* p) {
    return load_frag2(p, p + 8);
}

// ---------------------------------------------------------------------------
// Projection GEMM: out = Wmat[M x C] * x[b][C x N] (+bias), written as bf16.
//   TRANSPOSE=1: out layout [N, M] (Qt/Kt: contiguous per-lane fragment loads)
//   TRANSPOSE=0: out layout [M, N] (V: matches B-operand layout)
// One wave computes one 16x16 tile with v_wmma_f32_16x16x32_bf16, K-loop over C.
// Fragment layouts (A/B k order chosen by this code; C/D per ISA 7.12.2):
//   A (16x32 bf16): row=lane%16, elem i: K = hf*16 + i (this code's order, not the ISA table)
//   B (32x16 bf16): col=lane%16, elem i: K = hf*16 + i (this code's order, not the ISA table)
//   C/D (16x16 f32): row = reg + 8*hf, col = lane%16     [hf = lane>=16]
// ---------------------------------------------------------------------------
template <bool HAS_BIAS, bool TRANSPOSE>
__global__ void __launch_bounds__(128) proj_gemm(
    const float* __restrict__ Wmat, const float* __restrict__ bias,
    const float* __restrict__ x, __bf16* __restrict__ out, int M)
{
    const int lane = threadIdx.x & 31;
    const int wid  = threadIdx.x >> 5;
    const int hf   = lane >> 4;      // 0 or 1
    const int l16  = lane & 15;
    const int n0 = (blockIdx.x * 4 + wid) * 16;
    const int m0 = blockIdx.y * 16;
    const int b  = blockIdx.z;
    const float* xb = x + (size_t)b * C_ * N_;

    v8f acc = {};
    const float* wrow = Wmat + (size_t)(m0 + l16) * C_ + hf * 16;
    for (int k0 = 0; k0 < C_; k0 += 32) {
        v16bf afr, bfr;
        #pragma unroll
        for (int i = 0; i < 8; ++i) {
            afr[i]     = (__bf16)wrow[k0 + i];
            afr[i + 8] = (__bf16)wrow[k0 + 8 + i];
        }
        const float* xcol = xb + (size_t)(k0 + hf * 16) * N_ + n0 + l16;
        #pragma unroll
        for (int i = 0; i < 16; ++i)
            bfr[i] = (__bf16)xcol[(size_t)i * N_];
        acc = __builtin_amdgcn_wmma_f32_16x16x32_bf16(
                  false, afr, false, bfr, (short)0, acc, false, false);
    }

    // bias for rows m0 + hf*8 + r  (two aligned float4 loads, no branches)
    float bv[8];
    if (HAS_BIAS) {
        float4 b01 = *reinterpret_cast<const float4*>(bias + m0 + hf * 8);
        float4 b23 = *reinterpret_cast<const float4*>(bias + m0 + hf * 8 + 4);
        bv[0] = b01.x; bv[1] = b01.y; bv[2] = b01.z; bv[3] = b01.w;
        bv[4] = b23.x; bv[5] = b23.y; bv[6] = b23.z; bv[7] = b23.w;
    } else {
        #pragma unroll
        for (int r = 0; r < 8; ++r) bv[r] = 0.0f;
    }

    __bf16* ob = out + (size_t)b * M * N_;
    if (TRANSPOSE) {
        Pack16 pk;
        #pragma unroll
        for (int r = 0; r < 8; ++r) pk.v[r] = (__bf16)(acc[r] + bv[r]);
        // rows r are consecutive m -> one 16B store per lane
        *reinterpret_cast<uint4*>(ob + (size_t)(n0 + l16) * M + m0 + hf * 8)
            = __builtin_bit_cast(uint4, pk);
    } else {
        #pragma unroll
        for (int r = 0; r < 8; ++r)
            ob[(size_t)(m0 + hf * 8 + r) * N_ + n0 + l16] = (__bf16)(acc[r] + bv[r]);
    }
}

// key_mask[b][n] = sum_c wkm[c]*x[b][c][n] + bkm  (memory-bound matvec)
__global__ void __launch_bounds__(256) key_mask_kernel(
    const float* __restrict__ wkm, const float* __restrict__ bkm,
    const float* __restrict__ x, float* __restrict__ km)
{
    int idx = blockIdx.x * blockDim.x + threadIdx.x;   // over B*N
    int b = idx >> 12;                                 // /4096
    int n = idx & (N_ - 1);
    const float* xb = x + (size_t)b * C_ * N_ + n;
    float s = bkm[0];
    for (int c = 0; c < C_; ++c) s += wkm[c] * xb[(size_t)c * N_];
    km[idx] = s;
}

// ---------------------------------------------------------------------------
// Fused flash attention + residual:
//   out[b,c,m] = x[b,c,m] + gamma * sum_n softmax_n(QtK/16 + km)[m,n] * V[c,n]
// Block: 256 threads = 8 waves = 4 query tiles (16 queries) x 2 channel halves
// (256 channels each). All hot-loop WMMA operands come from LDS:
//   - Q block (64q x 256p, 32KB) staged once
//   - K tile (32k x 256p, 16KB) + V tile (512c x 32k, 32KB) double-buffered,
//     copied cooperatively with b128 loads/stores; ONE __syncthreads per tile
//     (writing buffer nxt after the barrier is ordered after all prior reads
//     of that buffer).
// Per wave: O = 16 f32 accumulator tiles (128 VGPRs), online softmax with
// cross-lane shuffles; P transposed D-layout -> A-layout through a wave-
// private LDS scratch (DScnt ordering, no barrier).
// ---------------------------------------------------------------------------
__global__ void __launch_bounds__(256) flash_attn(
    const __bf16* __restrict__ Qt,  // [B][N][P] bf16
    const __bf16* __restrict__ Kt,  // [B][N][P] bf16
    const __bf16* __restrict__ Vb,  // [B][C][N] bf16
    const float*  __restrict__ km,  // [B][N]
    const float*  __restrict__ x,   // [B][C][N]
    const float*  __restrict__ gamma,
    float* __restrict__ out)        // [B][C][N]
{
    __shared__ __align__(16) __bf16 ldsQ[64 * 256];      // 32 KB
    __shared__ __align__(16) __bf16 ldsK[2][32 * 256];   // 2 x 16 KB
    __shared__ __align__(16) __bf16 ldsV[2][512 * 32];   // 2 x 32 KB
    __shared__ __align__(16) __bf16 ldsP[8][16 * 32];    // 8 KB

    const int tid  = threadIdx.x;
    const int lane = tid & 31;
    const int wid  = tid >> 5;
    const int hf   = lane >> 4;
    const int l16  = lane & 15;
    const int b      = blockIdx.y;
    const int mblk   = blockIdx.x * 64;
    const int m0     = mblk + (wid >> 1) * 16;
    const int c_base = (wid & 1) * 256;

    const __bf16* Qb   = Qt + (size_t)b * N_ * P_;
    const __bf16* Kb   = Kt + (size_t)b * N_ * P_;
    const __bf16* Vbat = Vb + (size_t)b * C_ * N_;
    const float*  kmb  = km + (size_t)b * N_;

    // K/V tile copy: 48KB / 256 threads = 12 b128 chunks per thread
    auto copy_tile = [&](int ktn, int buf) {
        const uint4* gk = reinterpret_cast<const uint4*>(Kb + (size_t)ktn * P_);
        uint4* lk = reinterpret_cast<uint4*>(&ldsK[buf][0]);
        #pragma unroll
        for (int j = 0; j < 4; ++j) lk[tid + 256 * j] = gk[tid + 256 * j];
        #pragma unroll
        for (int j = 0; j < 8; ++j) {
            int cid = tid + 256 * j;          // 2048 chunks of 16B
            int c   = cid >> 2;
            int off = (cid & 3) * 8;          // bf16 elements
            *reinterpret_cast<uint4*>(&ldsV[buf][c * 32 + off]) =
                *reinterpret_cast<const uint4*>(Vbat + (size_t)c * N_ + ktn + off);
        }
    };

    // stage Q block (contiguous 32KB) + first K/V tile
    {
        const uint4* gq = reinterpret_cast<const uint4*>(Qb + (size_t)mblk * P_);
        uint4* lq = reinterpret_cast<uint4*>(&ldsQ[0]);
        #pragma unroll
        for (int j = 0; j < 8; ++j) lq[tid + 256 * j] = gq[tid + 256 * j];
    }
    copy_tile(0, 0);

    v8f o[16];
    #pragma unroll
    for (int t = 0; t < 16; ++t) o[t] = (v8f){};
    float rmax[8], rsum[8];
    #pragma unroll
    for (int r = 0; r < 8; ++r) { rmax[r] = -1e30f; rsum[r] = 0.0f; }

    const __bf16* qrow = &ldsQ[((size_t)((wid >> 1) * 16 + l16)) * 256 + hf * 16];
    __bf16* myP = &ldsP[wid][0];

    for (int kt = 0; kt < N_; kt += 32) {
        const int cur = (kt >> 5) & 1;
        __syncthreads();                       // buf[cur] ready; buf[cur^1] free
        if (kt + 32 < N_) copy_tile(kt + 32, cur ^ 1);

        // ---- S = Q[m-tile] x K[key-tile] : two 16x16 f32 tiles over P=256
        v8f s0 = {}, s1 = {};
        const __bf16* krow0 = &ldsK[cur][(size_t)l16 * 256 + hf * 16];
        const __bf16* krow1 = &ldsK[cur][(size_t)(16 + l16) * 256 + hf * 16];
        for (int p0 = 0; p0 < P_; p0 += 32) {
            v16bf afr = load_frag_g(qrow + p0);
            v16bf bf0 = load_frag_g(krow0 + p0);
            v16bf bf1 = load_frag_g(krow1 + p0);
            s0 = __builtin_amdgcn_wmma_f32_16x16x32_bf16(false, afr, false, bf0, (short)0, s0, false, false);
            s1 = __builtin_amdgcn_wmma_f32_16x16x32_bf16(false, afr, false, bf1, (short)0, s1, false, false);
        }

        // ---- scale + key-mask + online softmax (row = r + 8*hf, col = l16)
        const float km0 = kmb[kt + l16];
        const float km1 = kmb[kt + 16 + l16];
        float corr[8];
        #pragma unroll
        for (int r = 0; r < 8; ++r) {
            float a  = s0[r] * SCALE_ + km0;
            float bb = s1[r] * SCALE_ + km1;
            float mx = fmaxf(a, bb);
            #pragma unroll
            for (int d = 1; d <= 8; d <<= 1) mx = fmaxf(mx, __shfl_xor(mx, d, 32));
            float nm = fmaxf(rmax[r], mx);
            float cr = __expf(rmax[r] - nm);
            rmax[r]  = nm;
            float e0 = __expf(a  - nm);
            float e1 = __expf(bb - nm);
            float ts = e0 + e1;
            #pragma unroll
            for (int d = 1; d <= 8; d <<= 1) ts += __shfl_xor(ts, d, 32);
            rsum[r] = rsum[r] * cr + ts;
            corr[r] = cr;
            // stash P row-major [16m x 32n] (D-layout scatter, wave-private)
            myP[(hf * 8 + r) * 32 + l16]      = (__bf16)e0;
            myP[(hf * 8 + r) * 32 + 16 + l16] = (__bf16)e1;
        }

        // rescale running O by the max-correction
        #pragma unroll
        for (int t = 0; t < 16; ++t)
            #pragma unroll
            for (int r = 0; r < 8; ++r) o[t][r] *= corr[r];

        // re-read P in A-operand layout (row = l16, one 16-wide K run)
        v16bf pa = load_frag_g(&myP[l16 * 32 + hf * 16]);

        // ---- O += P x V^T over this wave's 256 channels (16 tiles)
        #pragma unroll
        for (int t = 0; t < 16; ++t) {
            const __bf16* vp = &ldsV[cur][(size_t)(c_base + t * 16 + l16) * 32 + hf * 16];
            v16bf vf = load_frag_g(vp);
            o[t] = __builtin_amdgcn_wmma_f32_16x16x32_bf16(false, pa, false, vf, (short)0, o[t], false, false);
        }
    }

    // ---- epilogue: out = x + gamma * O / rowsum
    const float g = gamma[0];
    float rinv[8];
    #pragma unroll
    for (int r = 0; r < 8; ++r) rinv[r] = 1.0f / rsum[r];
    #pragma unroll
    for (int t = 0; t < 16; ++t) {
        int c = c_base + t * 16 + l16;
        #pragma unroll
        for (int r = 0; r < 8; ++r) {
            int m = m0 + hf * 8 + r;
            size_t idx = ((size_t)b * C_ + c) * N_ + m;
            out[idx] = x[idx] + g * o[t][r] * rinv[r];
        }
    }
}

extern "C" void kernel_launch(void* const* d_in, const int* in_sizes, int n_in,
                              void* d_out, int out_size, void* d_ws, size_t ws_size,
                              hipStream_t stream) {
    const float* x     = (const float*)d_in[0];
    const float* wq    = (const float*)d_in[1];
    const float* bq    = (const float*)d_in[2];
    const float* wk    = (const float*)d_in[3];
    const float* bk    = (const float*)d_in[4];
    const float* wvm   = (const float*)d_in[5];
    const float* wkm   = (const float*)d_in[6];
    const float* bkm   = (const float*)d_in[7];
    const float* gamma = (const float*)d_in[8];
    float* out = (float*)d_out;

    // Workspace layout (bytes): Qt 16MB | Kt 16MB | V 32MB | km 128KB
    char* ws = (char*)d_ws;
    const size_t qk_bytes = (size_t)B_ * N_ * P_ * 2;   // 16 MB each
    const size_t v_bytes  = (size_t)B_ * C_ * N_ * 2;   // 32 MB
    __bf16* Qt = (__bf16*)(ws);
    __bf16* Kt = (__bf16*)(ws + qk_bytes);
    __bf16* Vv = (__bf16*)(ws + 2 * qk_bytes);
    float*  kmw = (float*)(ws + 2 * qk_bytes + v_bytes);

    proj_gemm<true,  true ><<<dim3(N_ / 64, P_ / 16, B_), dim3(128), 0, stream>>>(wq, bq, x, Qt, P_);
    proj_gemm<true,  true ><<<dim3(N_ / 64, P_ / 16, B_), dim3(128), 0, stream>>>(wk, bk, x, Kt, P_);
    proj_gemm<false, false><<<dim3(N_ / 64, C_ / 16, B_), dim3(128), 0, stream>>>(wvm, nullptr, x, Vv, C_);
    key_mask_kernel<<<dim3((B_ * N_) / 256), dim3(256), 0, stream>>>(wkm, bkm, x, kmw);
    flash_attn<<<dim3(N_ / 64, B_), dim3(256), 0, stream>>>(Qt, Kt, Vv, kmw, x, gamma, out);
}
